// MultiTaskGATSoilClassifier_17549236372084
// MI455X (gfx1250) — matrix operand order repaired and checked by reading, not compiled, not run on hardware
//
#include <hip/hip_runtime.h>
#include <hip/hip_bf16.h>
#include <math.h>
#include <stdint.h>

// ---------------------------------------------------------------------------
// Types for CDNA5 WMMA
// ---------------------------------------------------------------------------
typedef __attribute__((ext_vector_type(16))) __bf16 v16bf;
typedef __attribute__((ext_vector_type(8)))  float  v8f;

#define TPB 256

// ---------------------------------------------------------------------------
// Elementwise / utility kernels
// ---------------------------------------------------------------------------
__global__ void k_fill_f32(float* __restrict__ p, float v, int n) {
  int i = blockIdx.x * blockDim.x + threadIdx.x;
  if (i < n) p[i] = v;
}

__global__ void k_build_edges(const int* __restrict__ ei,
                              int* __restrict__ src, int* __restrict__ dst,
                              int E, int Et) {
  int i = blockIdx.x * blockDim.x + threadIdx.x;
  if (i >= Et) return;
  if (i < E) { src[i] = ei[i]; dst[i] = ei[E + i]; }
  else       { src[i] = i - E; dst[i] = i - E;    }
}

__global__ void k_cvt_bf16(const float* __restrict__ in,
                           __bf16* __restrict__ out, int n) {
  int i = blockIdx.x * blockDim.x + threadIdx.x;
  if (i < n) out[i] = (__bf16)in[i];
}

// W[K x Nc] (row-major, fp32) -> Wt[Npad x K] (row-major bf16, transposed,
// zero-padded columns so Npad is a multiple of 32)
__global__ void k_cvt_w_t(const float* __restrict__ W, __bf16* __restrict__ Wt,
                          int K, int Nc, int Npad) {
  int i = blockIdx.x * blockDim.x + threadIdx.x;
  int total = Npad * K;
  if (i >= total) return;
  int n = i / K, k = i - n * K;
  float v = (n < Nc) ? W[(long)k * Nc + n] : 0.f;
  Wt[(long)n * K + k] = (__bf16)v;
}

// ---------------------------------------------------------------------------
// bf16 WMMA GEMM:  C[M x Nc] = A[M x K] * B   (B given transposed: Bt[Nc x K])
// One 16x32 output tile per wave32 (two WMMAs sharing one A fragment).
// K multiple of 32, Nc multiple of 32, M multiple of 16.
// All fragment loads are 16-byte (global_load_b128).
// ---------------------------------------------------------------------------
__global__ void k_gemm_wmma_bf16(const __bf16* __restrict__ A,
                                 const __bf16* __restrict__ Bt,
                                 float* __restrict__ C,
                                 int M, int Nc, int K) {
  const int lane = threadIdx.x & 31;
  const int wave = threadIdx.x >> 5;
  const int ntn  = Nc >> 5;                   // 32-column tiles
  const int ntiles = (M >> 4) * ntn;
  int tile = blockIdx.x * (blockDim.x >> 5) + wave;
  if (tile >= ntiles) return;                 // wave-uniform; EXEC stays full
  const int tm   = tile / ntn;
  const int tn   = tile - tm * ntn;
  const int half = lane >> 4;                 // 0: lanes 0-15, 1: lanes 16-31
  const int l15  = lane & 15;
  const int row  = (tm << 4) + l15;           // A-matrix M index for this lane
  const int col0 = (tn << 5) + l15;           // first 16-col tile N index
  const int col1 = col0 + 16;                 // second 16-col tile N index
  const uint4* A4 = (const uint4*)A;          // 8 bf16 per uint4
  const uint4* B4 = (const uint4*)Bt;
  const long arow = (long)row  * K;
  const long brw0 = (long)col0 * K;
  const long brw1 = (long)col1 * K;

  v8f acc0 = {}, acc1 = {};
  for (int k0 = 0; k0 < K; k0 += 32) {
    union { v16bf v; uint4 q[2]; } af, bf0, bf1;
    // A 16x32 bf16 fragment (ISA 7.12.2):
    //   elements 0..7  <- K = k0 + half*8 + {0..7}   (contiguous, 16B)
    //   elements 8..15 <- K = k0 + 16 + half*8 + {0..7}
    af.q[0]  = A4[(arow + k0 +      (half << 3)) >> 3];
    af.q[1]  = A4[(arow + k0 + 16 + (half << 3)) >> 3];
    // B 32x16 fragment: same K order as A (half*8 + {0..7}, 16 + half*8 + {0..7})
    bf0.q[0] = B4[(brw0 + k0 +      (half << 3)) >> 3];
    bf0.q[1] = B4[(brw0 + k0 + 16 + (half << 3)) >> 3];
    bf1.q[0] = B4[(brw1 + k0 +      (half << 3)) >> 3];
    bf1.q[1] = B4[(brw1 + k0 + 16 + (half << 3)) >> 3];
    acc0 = __builtin_amdgcn_wmma_f32_16x16x32_bf16(
        false, af.v, false, bf0.v, (short)0, acc0, false, false);
    acc1 = __builtin_amdgcn_wmma_f32_16x16x32_bf16(
        false, af.v, false, bf1.v, (short)0, acc1, false, false);
  }
  // C/D layout: VGPR r -> M = tm*16 + r + half*8, N = col
#pragma unroll
  for (int r = 0; r < 8; ++r) {
    int m = (tm << 4) + r + (half << 3);
    C[(long)m * Nc + col0] = acc0[r];
    C[(long)m * Nc + col1] = acc1[r];
  }
}

// ---------------------------------------------------------------------------
// Attention kernels
// ---------------------------------------------------------------------------
// alpha_s[n,h] = sum_c feat[n, h*C+c] * a_s[h*C+c]   (ld = padded feature row)
__global__ void k_calc_alpha(const float* __restrict__ feat,
                             const float* __restrict__ aw_s,
                             const float* __restrict__ aw_d,
                             float* __restrict__ as_o, float* __restrict__ ad_o,
                             int N, int H, int C, int ld) {
  int i = blockIdx.x * blockDim.x + threadIdx.x;
  if (i >= N * H) return;
  int n = i / H, hh = i - n * H;
  const float* fp = feat + (long)n * ld + hh * C;
  const float* ws = aw_s + hh * C;
  const float* wd = aw_d + hh * C;
  float s = 0.f, d = 0.f;
  for (int c = 0; c < C; ++c) { float v = fp[c]; s += v * ws[c]; d += v * wd[c]; }
  as_o[i] = s; ad_o[i] = d;
}

__device__ __forceinline__ void atomicMaxFloat(float* addr, float v) {
  if (v >= 0.f) atomicMax((int*)addr, __float_as_int(v));
  else          atomicMin((unsigned int*)addr, __float_as_uint(v));
}

// pass1: e = leaky_relu(as[src]+ad[dst]); m[dst] = max(m[dst], e)
__global__ void k_edge_pass1(const int* __restrict__ src, const int* __restrict__ dst,
                             const float* __restrict__ as_, const float* __restrict__ ad_,
                             float* __restrict__ e, float* __restrict__ m,
                             int Et, int H) {
  int i = blockIdx.x * blockDim.x + threadIdx.x;
  if (i >= Et * H) return;
  int ed = i / H, hh = i - ed * H;
  int s = src[ed], d = dst[ed];
  float v = as_[(long)s * H + hh] + ad_[(long)d * H + hh];
  v = (v >= 0.f) ? v : 0.2f * v;
  e[i] = v;
  atomicMaxFloat(&m[(long)d * H + hh], v);
}

// pass2: ex = exp(e - m[dst]); denom[dst] += ex (in-place overwrite of e)
__global__ void k_edge_pass2(const int* __restrict__ dst,
                             float* __restrict__ e, const float* __restrict__ m,
                             float* __restrict__ denom, int Et, int H) {
  int i = blockIdx.x * blockDim.x + threadIdx.x;
  if (i >= Et * H) return;
  int ed = i / H, hh = i - ed * H;
  int d = dst[ed];
  float ex = expf(e[i] - m[(long)d * H + hh]);
  e[i] = ex;
  atomicAdd(&denom[(long)d * H + hh], ex);
}

// pass3: acc[dst, h, :] += (ex/denom[dst,h]) * feat[src, h, :]
__global__ void k_edge_pass3(const int* __restrict__ src, const int* __restrict__ dst,
                             const float* __restrict__ e, const float* __restrict__ denom,
                             const float* __restrict__ feat, float* __restrict__ acc,
                             int Et, int H, int C, int ld) {
  int i = blockIdx.x * blockDim.x + threadIdx.x;
  if (i >= Et * H) return;
  int ed = i / H, hh = i - ed * H;
  int s = src[ed], d = dst[ed];
  float coef = e[i] / (denom[(long)d * H + hh] + 1e-16f);
  const float* fs = feat + (long)s * ld + hh * C;
  float* ao = acc + (long)d * ld + hh * C;
  for (int c = 0; c < C; ++c) atomicAdd(&ao[c], coef * fs[c]);
}

// layer-1 post: h = elu(acc + b) -> bf16 (next GEMM's A)
__global__ void k_post_elu_bf16(const float* __restrict__ acc, const float* __restrict__ b,
                                __bf16* __restrict__ outb, int N, int D) {
  int i = blockIdx.x * blockDim.x + threadIdx.x;
  if (i >= N * D) return;
  float v = acc[i] + b[i % D];
  v = (v > 0.f) ? v : expm1f(v);
  outb[i] = (__bf16)v;
}

// layer-2 post: h = acc + b -> bf16
__global__ void k_post_bf16(const float* __restrict__ acc, const float* __restrict__ b,
                            __bf16* __restrict__ outb, int N, int D) {
  int i = blockIdx.x * blockDim.x + threadIdx.x;
  if (i >= N * D) return;
  outb[i] = (__bf16)(acc[i] + b[i % D]);
}

// head post: out[n, :] = log_softmax(acc[n, :nc] + b)
__global__ void k_post_head(const float* __restrict__ acc, const float* __restrict__ b,
                            float* __restrict__ out, int N, int nc, int ldp) {
  int n = blockIdx.x * blockDim.x + threadIdx.x;
  if (n >= N) return;
  const float* ap = acc + (long)n * ldp;
  float mx = -INFINITY;
  for (int c = 0; c < nc; ++c) { float v = ap[c] + b[c]; mx = v > mx ? v : mx; }
  float s = 0.f;
  for (int c = 0; c < nc; ++c) s += expf(ap[c] + b[c] - mx);
  float ls = mx + logf(s);
  float* op = out + (long)n * nc;
  for (int c = 0; c < nc; ++c) op[c] = ap[c] + b[c] - ls;
}

// ---------------------------------------------------------------------------
// Host orchestration
// ---------------------------------------------------------------------------
static inline unsigned gdim(long n) { return (unsigned)((n + TPB - 1) / TPB); }

extern "C" void kernel_launch(void* const* d_in, const int* in_sizes, int n_in,
                              void* d_out, int out_size, void* d_ws, size_t ws_size,
                              hipStream_t stream) {
  (void)in_sizes; (void)n_in; (void)out_size; (void)ws_size;

  constexpr int N = 50000, E = 800000, Et = E + N;
  constexpr int FIN = 128, H = 8, C = 32, DH = 256;

  const float* x  = (const float*)d_in[0];
  const int*   ei = (const int*)d_in[1];
  const float* W1  = (const float*)d_in[2];
  const float* a1s = (const float*)d_in[3];
  const float* a1d = (const float*)d_in[4];
  const float* b1  = (const float*)d_in[5];
  const float* W2  = (const float*)d_in[6];
  const float* a2s = (const float*)d_in[7];
  const float* a2d = (const float*)d_in[8];
  const float* b2  = (const float*)d_in[9];

  // task heads: W, a_s, a_d, b starting at index 10, 4 tensors each
  const int   head_nc[4]  = {12, 30, 60, 120};
  const int   head_ncp[4] = {32, 32, 64, 128};   // padded to multiple of 32
  long        head_off[4];
  head_off[0] = 0;
  head_off[1] = (long)N * 12;
  head_off[2] = (long)N * (12 + 30);
  head_off[3] = (long)N * (12 + 30 + 60);

  float* out = (float*)d_out;

  // ---- carve workspace -----------------------------------------------------
  size_t off = 0;
  auto carve = [&](size_t bytes) -> void* {
    void* p = (char*)d_ws + off;
    off = (off + bytes + 255) & ~(size_t)255;
    return p;
  };
  int*    sSrc = (int*)   carve((size_t)Et * 4);
  int*    sDst = (int*)   carve((size_t)Et * 4);
  __bf16* Abuf = (__bf16*)carve((size_t)N * DH * 2);   // bf16 GEMM input
  __bf16* Wt   = (__bf16*)carve((size_t)DH * DH * 2);  // bf16 transposed weight
  float*  g    = (float*) carve((size_t)N * DH * 4);   // GEMM output (features)
  float*  acc  = (float*) carve((size_t)N * DH * 4);   // aggregation target
  float*  aS   = (float*) carve((size_t)N * H * 4);
  float*  aD   = (float*) carve((size_t)N * H * 4);
  float*  mbuf = (float*) carve((size_t)N * H * 4);
  float*  dbuf = (float*) carve((size_t)N * H * 4);
  float*  ebuf = (float*) carve((size_t)Et * H * 4);

  // ---- edges + x -> bf16 ---------------------------------------------------
  k_build_edges<<<gdim(Et), TPB, 0, stream>>>(ei, sSrc, sDst, E, Et);
  k_cvt_bf16<<<gdim((long)N * FIN), TPB, 0, stream>>>(x, Abuf, N * FIN);

  // generic GAT layer driver
  auto gat_layer = [&](const float* W, int K, int Nc, int Npad,
                       const float* aws, const float* awd,
                       int Hh, int Cc) {
    // weight convert + transpose (+pad)
    k_cvt_w_t<<<gdim((long)Npad * K), TPB, 0, stream>>>(W, Wt, K, Nc, Npad);
    // GEMM: g = A(bf16) * W(bf16)   [N x Npad], 16x32 tile per wave
    long ntiles = (long)(N >> 4) * (Npad >> 5);
    k_gemm_wmma_bf16<<<(unsigned)((ntiles + 7) / 8), TPB, 0, stream>>>(
        Abuf, Wt, g, N, Npad, K);
    // attention coefficients
    k_calc_alpha<<<gdim((long)N * Hh), TPB, 0, stream>>>(
        g, aws, awd, aS, aD, N, Hh, Cc, Npad);
    // init reductions
    k_fill_f32<<<gdim((long)N * Hh), TPB, 0, stream>>>(mbuf, -INFINITY, N * Hh);
    k_fill_f32<<<gdim((long)N * Hh), TPB, 0, stream>>>(dbuf, 0.f, N * Hh);
    k_fill_f32<<<gdim((long)N * Npad), TPB, 0, stream>>>(acc, 0.f, N * Npad);
    // segment softmax + scatter aggregation
    k_edge_pass1<<<gdim((long)Et * Hh), TPB, 0, stream>>>(sSrc, sDst, aS, aD, ebuf, mbuf, Et, Hh);
    k_edge_pass2<<<gdim((long)Et * Hh), TPB, 0, stream>>>(sDst, ebuf, mbuf, dbuf, Et, Hh);
    k_edge_pass3<<<gdim((long)Et * Hh), TPB, 0, stream>>>(sSrc, sDst, ebuf, dbuf, g, acc,
                                                          Et, Hh, Cc, Npad);
  };

  // ---- layer 1: x[N,128] -> h1[N,256], ELU --------------------------------
  gat_layer(W1, FIN, DH, DH, a1s, a1d, H, C);
  k_post_elu_bf16<<<gdim((long)N * DH), TPB, 0, stream>>>(acc, b1, Abuf, N, DH);

  // ---- layer 2: h1[N,256] -> h2[N,256] ------------------------------------
  gat_layer(W2, DH, DH, DH, a2s, a2d, H, C);
  k_post_bf16<<<gdim((long)N * DH), TPB, 0, stream>>>(acc, b2, Abuf, N, DH);

  // ---- task heads: h2[N,256] -> logits -> log_softmax ---------------------
  for (int t = 0; t < 4; ++t) {
    const float* Wh = (const float*)d_in[10 + 4 * t + 0];
    const float* hs = (const float*)d_in[10 + 4 * t + 1];
    const float* hd = (const float*)d_in[10 + 4 * t + 2];
    const float* hb = (const float*)d_in[10 + 4 * t + 3];
    int nc = head_nc[t], ncp = head_ncp[t];
    gat_layer(Wh, DH, nc, ncp, hs, hd, /*H=*/1, /*C=*/nc);
    k_post_head<<<gdim(N), TPB, 0, stream>>>(acc, hb, out + head_off[t], N, nc, ncp);
  }
}
